// PipelineGPT_33251636806134
// MI455X (gfx1250) — matrix operand order repaired and checked by reading, not compiled, not run on hardware
//
#include <hip/hip_runtime.h>

// ---------------------------------------------------------------------------
// GPT-2-like forward (V=50304,H=1600,NH=32,HD=50,L=4,MLP=6400,B=2,S=1024)
// CDNA5 / gfx1250, wave32. WMMA f16->f32 GEMMs, TDM (tensor_load_to_lds)
// double-buffered tile staging, flash attention with WMMA.
// ---------------------------------------------------------------------------

typedef __attribute__((ext_vector_type(16))) _Float16 v16h;
typedef __attribute__((ext_vector_type(8)))  float    v8f;
typedef __attribute__((ext_vector_type(4)))  unsigned int u32x4;
typedef __attribute__((ext_vector_type(8)))  int      i32x8;
typedef __attribute__((ext_vector_type(4)))  int      i32x4;

#define DEVINL __device__ __forceinline__

constexpr int Vv   = 50304;
constexpr int Hh   = 1600;
constexpr int NHh  = 32;
constexpr int HDd  = 50;
constexpr int Ll   = 4;
constexpr int MLPd = 6400;
constexpr int Bb   = 2;
constexpr int Ss   = 1024;
constexpr int BS   = Bb * Ss;   // 2048 rows

DEVINL v8f wmma_f16(v16h a, v16h b, v8f c) {
  // D = A(16x32 f16) * B(32x16 f16) + C(16x16 f32)
  return __builtin_amdgcn_wmma_f32_16x16x32_f16(
      /*neg_a=*/false, a, /*neg_b=*/false, b,
      /*c_mod=*/(short)0, c, /*reuse_a=*/false, /*reuse_b=*/false);
}

// A-fragment (16x32 f16) from a row-major LDS tile. `row` points at the start
// of this lane's matrix row (lane m = lane&15). Contiguous k order used here (not the ISA table order):
// lanes 0-15: v0..7 = K0..15; lanes 16-31: K16..31 (contiguous, matches B).
DEVINL v16h frag_a16(const _Float16* row, int kstep, int hsel) {
  union { v16h v; unsigned u[8]; } f;
  const unsigned* p = (const unsigned*)row;   // pairs (2*f16) per u32
  const int b0 = kstep * 16 + 8 * hsel;
#pragma unroll
  for (int i = 0; i < 4; ++i) { f.u[i] = p[b0 + i]; f.u[4 + i] = p[b0 + 4 + i]; }
  return f.v;
}

// B-fragment (32x16 f16): lane holds column n=lane&15; element e <-> K=16*half+e,
// i.e. 16 contiguous f16 along the contraction dim (tile stored [N][K]).
DEVINL v16h frag_b16(const _Float16* p16) {
  union { v16h v; uint4 q[2]; } f;
  f.q[0] = *(const uint4*)(p16);
  f.q[1] = *(const uint4*)(p16 + 8);
  return f.v;
}

DEVINL float gelu_f(float x) {
  return 0.5f * x * (1.0f + erff(x * 0.70710678118654752f));
}

// Generic->LDS byte offset: ISA 10.2, LDS generic addresses carry the
// wave-relative LDS offset in addr[31:0].
DEVINL unsigned lds_off(const void* p) { return (unsigned)(size_t)p; }

// ---------------------------------------------------------------------------
// Tensor Data Mover: async 2D tile (width 64 f16 elements x `tile_rows` rows,
// row stride `stride_elems`) from global -> LDS. Rows >= rows_remaining are
// zero-filled by TDM OOB handling (replaces edge guards). ISA ch.8 D# layout.
// ---------------------------------------------------------------------------
DEVINL void tdm_load_tile_f16(unsigned lds_byte, const void* gaddr,
                              unsigned tile_rows, unsigned long long stride_elems,
                              unsigned rows_remaining) {
  const unsigned long long ga = (unsigned long long)(size_t)gaddr;
  u32x4 g0;
  g0[0] = 1u;                                     // count=1, user descriptor
  g0[1] = lds_byte;                               // lds_addr
  g0[2] = (unsigned)(ga & 0xFFFFFFFFull);         // global_addr[31:0]
  g0[3] = (unsigned)((ga >> 32) & 0x1FFFFFFull)   // global_addr[56:32]
          | (2u << 30);                           // type = 2 ("image")
  const unsigned td0 = 64u;                       // tensor_dim0 == tile width
  const unsigned td1 = rows_remaining;            // tensor_dim1 (OOB -> zero)
  i32x8 g1;
  g1[0] = (int)(1u << 16);                        // data_size=1 (2 bytes/elem)
  g1[1] = (int)((td0 & 0xFFFFu) << 16);           // tensor_dim0[15:0]
  g1[2] = (int)((td0 >> 16) | ((td1 & 0xFFFFu) << 16));
  g1[3] = (int)((td1 >> 16) | (64u << 16));       // tile_dim0 = 64
  g1[4] = (int)(tile_rows & 0xFFFFu);             // tile_dim1 (tile_dim2 = 0)
  g1[5] = (int)(stride_elems & 0xFFFFFFFFull);    // tensor_dim0_stride[31:0]
  g1[6] = (int)((stride_elems >> 32) & 0xFFFFull);// stride hi; dim1_stride lo=0
  g1[7] = 0;
  const i32x4 z4 = {0, 0, 0, 0};                  // groups 2/3 unused (2-D)
  const i32x8 z8 = {0, 0, 0, 0, 0, 0, 0, 0};
  __builtin_amdgcn_tensor_load_to_lds(g0, g1, z4, z4, z8, 0);
}

// ---------------------------------------------------------------------------
// fp32 [K,N] -> f16 [N,K] transpose-convert (tiled through LDS)
// ---------------------------------------------------------------------------
__global__ __launch_bounds__(256) void cvtT_kernel(const float* __restrict__ src,
                                                   _Float16* __restrict__ dst,
                                                   int K, int N) {
  __shared__ float tile[32][33];
  const int nb = blockIdx.x * 32, kb = blockIdx.y * 32;
  const int tx = threadIdx.x & 31, ty = threadIdx.x >> 5;
  for (int i = ty; i < 32; i += 8) {
    const int kk = kb + i, nn = nb + tx;
    tile[i][tx] = (kk < K && nn < N) ? src[(size_t)kk * N + nn] : 0.0f;
  }
  __syncthreads();
  for (int i = ty; i < 32; i += 8) {
    const int nn = nb + i, kk = kb + tx;
    if (nn < N && kk < K) dst[(size_t)nn * K + kk] = (_Float16)tile[tx][i];
  }
}

// ---------------------------------------------------------------------------
// embedding: x[row,col] = word_emb[ids[row]][col]*sqrt(H) + pos_emb[row%S][col]
// ---------------------------------------------------------------------------
__global__ __launch_bounds__(256) void embed_kernel(const int* __restrict__ ids,
                                                    const float* __restrict__ we,
                                                    const float* __restrict__ pe,
                                                    float* __restrict__ x) {
  const long long n = (long long)BS * Hh;
  for (long long i = (long long)blockIdx.x * 256 + threadIdx.x; i < n;
       i += (long long)gridDim.x * 256) {
    const int row = (int)(i / Hh), col = (int)(i % Hh);
    const int s = row % Ss;
    const int id = ids[row];
    x[i] = we[(size_t)id * Hh + col] * 40.0f /* sqrt(1600) */ +
           pe[(size_t)s * Hh + col];
  }
}

// ---------------------------------------------------------------------------
// LayerNorm over H=1600: one workgroup per row, fp32 in, f16 out
// ---------------------------------------------------------------------------
__global__ __launch_bounds__(256) void ln_kernel(const float* __restrict__ x,
                                                 const float* __restrict__ g,
                                                 const float* __restrict__ b,
                                                 _Float16* __restrict__ out) {
  const int row = blockIdx.x;
  const float* xr = x + (size_t)row * Hh;
  float s1 = 0.f, s2 = 0.f;
  for (int c = threadIdx.x; c < Hh; c += 256) { float t = xr[c]; s1 += t; s2 += t * t; }
#pragma unroll
  for (int off = 16; off >= 1; off >>= 1) {
    s1 += __shfl_xor(s1, off, 32);
    s2 += __shfl_xor(s2, off, 32);
  }
  __shared__ float r1[8], r2[8];
  if ((threadIdx.x & 31) == 0) { r1[threadIdx.x >> 5] = s1; r2[threadIdx.x >> 5] = s2; }
  __syncthreads();
  float t1 = 0.f, t2 = 0.f;
#pragma unroll
  for (int i = 0; i < 8; ++i) { t1 += r1[i]; t2 += r2[i]; }
  const float inv = 1.0f / (float)Hh;
  const float mu = t1 * inv;
  const float var = t2 * inv - mu * mu;
  const float rstd = rsqrtf(var + 1e-5f);
  _Float16* orow = out + (size_t)row * Hh;
  for (int c = threadIdx.x; c < Hh; c += 256)
    orow[c] = (_Float16)((xr[c] - mu) * rstd * g[c] + b[c]);
}

// ---------------------------------------------------------------------------
// WMMA GEMM with TDM double buffering:
//   C[M,N] = A[M,K](f16,row-major) @ W^T (W stored [N,K] f16) + epilogue
// Workgroup tile 64x128, 8 waves of 32x32, BK=64 (two 16x16x32 k-steps).
// Wave 0 issues tensor_load_to_lds for tile kb+1 into the alternate buffer
// while all waves compute on the current one; s_wait_tensorcnt gates reuse.
// EPI bits: 1 = f16 output, 2 = GELU, 4 = add fp32 residual (fp32 output)
// ---------------------------------------------------------------------------
template <int EPI>
__global__ __launch_bounds__(256) void gemm_wmma(const _Float16* __restrict__ A,
                                                 const _Float16* __restrict__ W,
                                                 const float* __restrict__ bias,
                                                 const float* resid, void* outp,
                                                 int M, int N, int K) {
  __shared__ __align__(16) _Float16 As[2][64][64];
  __shared__ __align__(16) _Float16 Bs[2][128][64];

  const int tid  = threadIdx.x;
  const int lane = tid & 31, wid = tid >> 5;
  const int hsel = lane >> 4, ln = lane & 15;
  const int wM = wid & 1, wN = wid >> 1;           // 2 x 4 wave grid
  const long long bM = (long long)blockIdx.y * 64;
  const long long bN = (long long)blockIdx.x * 128;

  const v8f zero8 = {0.f, 0.f, 0.f, 0.f, 0.f, 0.f, 0.f, 0.f};
  v8f acc[2][2];
#pragma unroll
  for (int i = 0; i < 2; ++i)
#pragma unroll
    for (int j = 0; j < 2; ++j) acc[i][j] = zero8;

  const _Float16* Abase = A + (size_t)bM * K;      // row bM of A
  const _Float16* Wbase = W + (size_t)bN * K;      // row bN of W^T
  const unsigned remM = (unsigned)(M - bM);        // grid guarantees > 0
  const unsigned remN = (unsigned)(N - bN);
  const int nkb = K >> 6;                          // K % 64 == 0 here

  if (wid == 0) {  // prologue: fetch tile 0 into buffer 0
    tdm_load_tile_f16(lds_off(&As[0][0][0]), Abase, 64u, (unsigned long long)K, remM);
    tdm_load_tile_f16(lds_off(&Bs[0][0][0]), Wbase, 128u, (unsigned long long)K, remN);
  }

  for (int kb = 0; kb < nkb; ++kb) {
    const int cur = kb & 1;
    if (wid == 0) {
      if (kb + 1 < nkb) {  // prefetch next tile into alternate buffer
        const _Float16* an = Abase + (size_t)(kb + 1) * 64;
        const _Float16* wn = Wbase + (size_t)(kb + 1) * 64;
        tdm_load_tile_f16(lds_off(&As[cur ^ 1][0][0]), an, 64u, (unsigned long long)K, remM);
        tdm_load_tile_f16(lds_off(&Bs[cur ^ 1][0][0]), wn, 128u, (unsigned long long)K, remN);
        __builtin_amdgcn_s_wait_tensorcnt(2);  // oldest pair (tile kb) done
      } else {
        __builtin_amdgcn_s_wait_tensorcnt(0);  // last tile done
      }
    }
    __syncthreads();  // tile `cur` visible to all waves

#pragma unroll
    for (int ks = 0; ks < 2; ++ks) {
      const v16h af0 = frag_a16(&As[cur][wM * 32 + ln][0],      ks, hsel);
      const v16h af1 = frag_a16(&As[cur][wM * 32 + 16 + ln][0], ks, hsel);
      const v16h bf0 = frag_b16(&Bs[cur][wN * 32 + ln][ks * 32 + 16 * hsel]);
      const v16h bf1 = frag_b16(&Bs[cur][wN * 32 + 16 + ln][ks * 32 + 16 * hsel]);
      acc[0][0] = wmma_f16(af0, bf0, acc[0][0]);
      acc[0][1] = wmma_f16(af0, bf1, acc[0][1]);
      acc[1][0] = wmma_f16(af1, bf0, acc[1][0]);
      acc[1][1] = wmma_f16(af1, bf1, acc[1][1]);
    }
    __syncthreads();  // all waves done reading `cur` before it is refilled
  }

  // epilogue: C layout -> row = r + 8*hsel (+tile), col = lane&15 (+tile)
#pragma unroll
  for (int tm = 0; tm < 2; ++tm)
#pragma unroll
    for (int tn = 0; tn < 2; ++tn) {
      const long long col = bN + wN * 32 + tn * 16 + ln;
      if (col >= N) continue;
      const float bvv = (bias != nullptr) ? bias[col] : 0.0f;
#pragma unroll
      for (int r = 0; r < 8; ++r) {
        const long long row = bM + wM * 32 + tm * 16 + r + 8 * hsel;
        if (row >= M) continue;
        float vv = acc[tm][tn][r] + bvv;
        if (EPI & 2) vv = gelu_f(vv);
        const size_t idx = (size_t)row * (size_t)N + (size_t)col;
        if (EPI & 1) {
          ((_Float16*)outp)[idx] = (_Float16)vv;
        } else {
          if (EPI & 4) vv += resid[idx];
          ((float*)outp)[idx] = vv;
        }
      }
    }
}

// ---------------------------------------------------------------------------
// Flash attention: grid = (S/64 q-blocks, B*NH). 128 threads = 4 waves,
// each wave owns 16 query rows. HD=50 zero-padded to 64. Causal blocks
// above the diagonal skipped. Online softmax matches reference's
// where(mask, s, -1e5) + softmax semantics.
// ---------------------------------------------------------------------------
__global__ __launch_bounds__(128) void attn_flash(const _Float16* __restrict__ q,
                                                  const _Float16* __restrict__ k,
                                                  const _Float16* __restrict__ v,
                                                  const int* __restrict__ amask,
                                                  _Float16* __restrict__ o) {
  __shared__ __align__(16) _Float16 Qs[64][64];      // q rows x d(pad)
  __shared__ __align__(16) _Float16 Ks[64][64];      // keys  x d(pad)
  __shared__ __align__(16) _Float16 Vs[64][64];      // d(pad) x keys (transposed)
  __shared__ __align__(16) _Float16 Ps[4][16][64];   // per-wave P staging

  const int tid = threadIdx.x, lane = tid & 31, wave = tid >> 5;
  const int hsel = lane >> 4, ln = lane & 15;
  const int qb = blockIdx.x;
  const int bidx = blockIdx.y / NHh, head = blockIdx.y % NHh;
  const int colbase = head * HDd;
  const float rscale = 0.1414213562373095f;  // 1/sqrt(50)

  {  // load Q tile
    const int r = tid >> 1, seg = tid & 1;
    const _Float16* src = q + ((size_t)(bidx * Ss + qb * 64 + r)) * Hh + colbase;
#pragma unroll
    for (int i = 0; i < 32; ++i) {
      const int d = seg * 32 + i;
      Qs[r][d] = (d < HDd) ? src[d] : (_Float16)0.0f;
    }
  }
  __syncthreads();
  v16h aq[2];
  aq[0] = frag_a16(&Qs[wave * 16 + ln][0], 0, hsel);
  aq[1] = frag_a16(&Qs[wave * 16 + ln][0], 1, hsel);

  const v8f zero8 = {0.f, 0.f, 0.f, 0.f, 0.f, 0.f, 0.f, 0.f};
  float mrow[8], lrow[8];
  v8f oacc[4];
#pragma unroll
  for (int r = 0; r < 8; ++r) { mrow[r] = -3.0e38f; lrow[r] = 0.0f; }
#pragma unroll
  for (int dt = 0; dt < 4; ++dt) oacc[dt] = zero8;

  for (int jb = 0; jb <= qb; ++jb) {
    __syncthreads();  // previous iteration done with Ks/Vs
    {                 // load K block and V block (transposed)
      const int r = tid >> 1, seg = tid & 1;
      const _Float16* sk = k + ((size_t)(bidx * Ss + jb * 64 + r)) * Hh + colbase;
      const _Float16* sv = v + ((size_t)(bidx * Ss + jb * 64 + r)) * Hh + colbase;
#pragma unroll
      for (int i = 0; i < 32; ++i) {
        const int d = seg * 32 + i;
        Ks[r][d] = (d < HDd) ? sk[d] : (_Float16)0.0f;
        Vs[d][r] = (d < HDd) ? sv[d] : (_Float16)0.0f;
      }
    }
    __syncthreads();

    // scores: Q(16x64) @ K^T(64x64) -> 4 tiles of 16 keys
    v8f sc[4];
#pragma unroll
    for (int nt = 0; nt < 4; ++nt) sc[nt] = zero8;
#pragma unroll
    for (int ks = 0; ks < 2; ++ks)
#pragma unroll
      for (int nt = 0; nt < 4; ++nt) {
        const v16h bf = frag_b16(&Ks[nt * 16 + ln][ks * 32 + 16 * hsel]);
        sc[nt] = wmma_f16(aq[ks], bf, sc[nt]);
      }

    int  jcol[4];
    bool kval[4];
#pragma unroll
    for (int nt = 0; nt < 4; ++nt) {
      jcol[nt] = jb * 64 + nt * 16 + ln;
      kval[nt] = (amask[bidx * Ss + jcol[nt]] != 0);
    }

    // online softmax per row (rows r+8*hsel live across lanes of one half)
#pragma unroll
    for (int r = 0; r < 8; ++r) {
      const int qrow = qb * 64 + wave * 16 + r + 8 * hsel;
      float rowmax = mrow[r];
      float sv4[4];
#pragma unroll
      for (int nt = 0; nt < 4; ++nt) {
        float s = sc[nt][r] * rscale;
        if (!(kval[nt] && (jcol[nt] <= qrow))) s = -100000.0f;
        sv4[nt] = s;
        rowmax = fmaxf(rowmax, s);
      }
#pragma unroll
      for (int off = 1; off < 16; off <<= 1)
        rowmax = fmaxf(rowmax, __shfl_xor(rowmax, off, 32));
      const float alpha = __expf(mrow[r] - rowmax);
      float ls = 0.f;
#pragma unroll
      for (int nt = 0; nt < 4; ++nt) {
        const float e = __expf(sv4[nt] - rowmax);
        sc[nt][r] = e;
        ls += e;
      }
#pragma unroll
      for (int off = 1; off < 16; off <<= 1) ls += __shfl_xor(ls, off, 32);
      lrow[r] = lrow[r] * alpha + ls;
      mrow[r] = rowmax;
#pragma unroll
      for (int dt = 0; dt < 4; ++dt) oacc[dt][r] *= alpha;
    }

    // re-layout P: C-fragment -> wave-private LDS -> A-fragment
#pragma unroll
    for (int nt = 0; nt < 4; ++nt)
#pragma unroll
      for (int r = 0; r < 8; ++r)
        Ps[wave][r + 8 * hsel][nt * 16 + ln] = (_Float16)sc[nt][r];
    asm volatile("s_wait_dscnt 0" ::: "memory");

    // O += P(16x64) @ V(64x64)
#pragma unroll
    for (int ks = 0; ks < 2; ++ks) {
      const v16h pa = frag_a16(&Ps[wave][ln][0], ks, hsel);
#pragma unroll
      for (int dt = 0; dt < 4; ++dt) {
        const v16h bf = frag_b16(&Vs[dt * 16 + ln][ks * 32 + 16 * hsel]);
        oacc[dt] = wmma_f16(pa, bf, oacc[dt]);
      }
    }
  }

  // write O / l
#pragma unroll
  for (int dt = 0; dt < 4; ++dt) {
    const int d = dt * 16 + ln;
    if (d >= HDd) continue;
#pragma unroll
    for (int r = 0; r < 8; ++r) {
      const size_t row = (size_t)(bidx * Ss + qb * 64 + wave * 16 + r + 8 * hsel);
      o[row * Hh + colbase + d] = (_Float16)(oacc[dt][r] / lrow[r]);
    }
  }
}

// ---------------------------------------------------------------------------
// Host orchestration
// ---------------------------------------------------------------------------
extern "C" void kernel_launch(void* const* d_in, const int* in_sizes, int n_in,
                              void* d_out, int out_size, void* d_ws, size_t ws_size,
                              hipStream_t stream) {
  (void)in_sizes; (void)n_in; (void)out_size; (void)ws_size;

  const int*   ids   = (const int*)d_in[0];
  const int*   am    = (const int*)d_in[1];
  const float* wemb  = (const float*)d_in[2];
  const float* pemb  = (const float*)d_in[3];
  const float* ln1w  = (const float*)d_in[4];
  const float* ln1b  = (const float*)d_in[5];
  const float* wq    = (const float*)d_in[6];
  const float* bq    = (const float*)d_in[7];
  const float* wk    = (const float*)d_in[8];
  const float* bk    = (const float*)d_in[9];
  const float* wv    = (const float*)d_in[10];
  const float* bv    = (const float*)d_in[11];
  const float* wo    = (const float*)d_in[12];
  const float* bo    = (const float*)d_in[13];
  const float* ln2w  = (const float*)d_in[14];
  const float* ln2b  = (const float*)d_in[15];
  const float* w1    = (const float*)d_in[16];
  const float* b1    = (const float*)d_in[17];
  const float* w2    = (const float*)d_in[18];
  const float* b2    = (const float*)d_in[19];
  const float* lnfw  = (const float*)d_in[20];
  const float* lnfb  = (const float*)d_in[21];
  const float* headw = (const float*)d_in[22];

  char* base = (char*)d_ws;
  size_t off = 0;
  auto allocH = [&](size_t n) -> _Float16* {
    _Float16* p = (_Float16*)(base + off);
    off += ((n * sizeof(_Float16)) + 255) & ~(size_t)255;
    return p;
  };
  auto allocF = [&](size_t n) -> float* {
    float* p = (float*)(base + off);
    off += ((n * sizeof(float)) + 255) & ~(size_t)255;
    return p;
  };

  _Float16* wqT = allocH((size_t)Ll * Hh * Hh);
  _Float16* wkT = allocH((size_t)Ll * Hh * Hh);
  _Float16* wvT = allocH((size_t)Ll * Hh * Hh);
  _Float16* woT = allocH((size_t)Ll * Hh * Hh);
  _Float16* w1T = allocH((size_t)Ll * MLPd * Hh);
  _Float16* w2T = allocH((size_t)Ll * Hh * MLPd);
  _Float16* hdT = allocH((size_t)Vv * Hh);
  _Float16* hb  = allocH((size_t)BS * Hh);
  _Float16* qb_ = allocH((size_t)BS * Hh);
  _Float16* kb_ = allocH((size_t)BS * Hh);
  _Float16* vb_ = allocH((size_t)BS * Hh);
  _Float16* ab_ = allocH((size_t)BS * Hh);
  _Float16* mb_ = allocH((size_t)BS * MLPd);
  float*    xb  = allocF((size_t)BS * Hh);

  auto cvt = [&](const float* s, _Float16* d, int K, int N) {
    dim3 g((N + 31) / 32, (K + 31) / 32);
    cvtT_kernel<<<g, 256, 0, stream>>>(s, d, K, N);
  };

  const size_t hh = (size_t)Hh * Hh, hm = (size_t)Hh * MLPd;
  for (int l = 0; l < Ll; ++l) {
    cvt(wq + l * hh, wqT + l * hh, Hh, Hh);
    cvt(wk + l * hh, wkT + l * hh, Hh, Hh);
    cvt(wv + l * hh, wvT + l * hh, Hh, Hh);
    cvt(wo + l * hh, woT + l * hh, Hh, Hh);
    cvt(w1 + l * hm, w1T + l * hm, Hh, MLPd);   // [H,MLP] -> [MLP,H]
    cvt(w2 + l * hm, w2T + l * hm, MLPd, Hh);   // [MLP,H] -> [H,MLP]
  }
  cvt(headw, hdT, Hh, Vv);                      // [H,V] -> [V,H]

  embed_kernel<<<4096, 256, 0, stream>>>(ids, wemb, pemb, xb);

  auto gemm = [&](const _Float16* Ain, const _Float16* Wt, const float* bias,
                  const float* resid, void* outp, int M, int N, int K, int epi) {
    dim3 g((N + 127) / 128, (M + 63) / 64);
    switch (epi) {
      case 0: gemm_wmma<0><<<g, 256, 0, stream>>>(Ain, Wt, bias, resid, outp, M, N, K); break;
      case 1: gemm_wmma<1><<<g, 256, 0, stream>>>(Ain, Wt, bias, resid, outp, M, N, K); break;
      case 3: gemm_wmma<3><<<g, 256, 0, stream>>>(Ain, Wt, bias, resid, outp, M, N, K); break;
      case 4: gemm_wmma<4><<<g, 256, 0, stream>>>(Ain, Wt, bias, resid, outp, M, N, K); break;
    }
  };

  for (int l = 0; l < Ll; ++l) {
    ln_kernel<<<BS, 256, 0, stream>>>(xb, ln1w + (size_t)l * Hh, ln1b + (size_t)l * Hh, hb);
    gemm(hb, wqT + l * hh, bq + (size_t)l * Hh, nullptr, qb_, BS, Hh, Hh, 1);
    gemm(hb, wkT + l * hh, bk + (size_t)l * Hh, nullptr, kb_, BS, Hh, Hh, 1);
    gemm(hb, wvT + l * hh, bv + (size_t)l * Hh, nullptr, vb_, BS, Hh, Hh, 1);
    attn_flash<<<dim3(Ss / 64, Bb * NHh), 128, 0, stream>>>(qb_, kb_, vb_, am, ab_);
    gemm(ab_, woT + l * hh, bo + (size_t)l * Hh, xb, xb, BS, Hh, Hh, 4);
    ln_kernel<<<BS, 256, 0, stream>>>(xb, ln2w + (size_t)l * Hh, ln2b + (size_t)l * Hh, hb);
    gemm(hb, w1T + l * hm, b1 + (size_t)l * MLPd, nullptr, mb_, BS, MLPd, Hh, 3);
    gemm(mb_, w2T + l * hm, b2 + (size_t)l * Hh, xb, xb, BS, Hh, MLPd, 4);
  }

  ln_kernel<<<BS, 256, 0, stream>>>(xb, lnfw, lnfb, hb);
  gemm(hb, hdT, nullptr, nullptr, d_out, BS, Vv, Hh, 0);  // logits fp32
}
